// RelGATModel_84542136254721
// MI455X (gfx1250) — matrix operand order repaired and checked by reading, not compiled, not run on hardware
//
#include <hip/hip_runtime.h>
#include <hip/hip_bf16.h>

typedef __attribute__((ext_vector_type(16))) __bf16 v16bf;
typedef __attribute__((ext_vector_type(8)))  __bf16 v8bf;
typedef __attribute__((ext_vector_type(8)))  float  v8f;

#define NEG_SLOPE 0.2f
#define EPSV 1e-16f
#define HH 4
#define DD 200
#define CC 800      // HH*DD
#define IN_DIM 256

// ---------------------------------------------------------------------------
// Split fp32 -> bf16 hi + bf16 lo (error compensation for WMMA GEMM)
// ---------------------------------------------------------------------------
__global__ void k_split_x(const float* __restrict__ x, __bf16* __restrict__ xh,
                          __bf16* __restrict__ xl, long total) {
  long i = (long)blockIdx.x * blockDim.x + threadIdx.x;
  if (i >= total) return;
  float v = x[i];
  __bf16 hi = (__bf16)v;
  float rem = v - (float)hi;
  xh[i] = hi;
  xl[i] = (__bf16)rem;
}

// W [K=256, N=800] row-major -> transposed bf16 hi/lo [800, 256]
__global__ void k_split_wT(const float* __restrict__ W, __bf16* __restrict__ wh,
                           __bf16* __restrict__ wl, int K, int Ncol) {
  int i = blockIdx.x * blockDim.x + threadIdx.x;
  if (i >= K * Ncol) return;
  int k = i / Ncol, n = i % Ncol;
  float v = W[i];
  __bf16 hi = (__bf16)v;
  float rem = v - (float)hi;
  wh[(long)n * K + k] = hi;
  wl[(long)n * K + k] = (__bf16)rem;
}

// ---------------------------------------------------------------------------
// h = x @ W via v_wmma_f32_16x16x32_bf16, one wave per 16x16 output tile.
// Error-compensated: acc += Ah*Bh + Ah*Bl + Al*Bh  (fp32-class accuracy)
// ---------------------------------------------------------------------------
__global__ void k_gemm_wmma(const __bf16* __restrict__ xh, const __bf16* __restrict__ xl,
                            const __bf16* __restrict__ wh, const __bf16* __restrict__ wl,
                            float* __restrict__ hout, int Nrows) {
  int gid  = blockIdx.x * blockDim.x + threadIdx.x;
  int wave = gid >> 5;
  int lane = gid & 31;
  int tilesM = (Nrows + 15) >> 4;
  int tilesN = CC / 16;                       // 50
  if (wave >= tilesM * tilesN) return;
  int tm = wave / tilesN, tn = wave % tilesN;
  int m0 = tm << 4, n0 = tn << 4;

  int lhi  = lane >> 4;                       // 0: lanes 0-15, 1: lanes 16-31
  int lmod = lane & 15;

  int arow = m0 + lmod; if (arow >= Nrows) arow = Nrows - 1;
  const __bf16* xhrow = xh + (long)arow * IN_DIM;
  const __bf16* xlrow = xl + (long)arow * IN_DIM;
  const __bf16* whcol = wh + (long)(n0 + lmod) * IN_DIM;
  const __bf16* wlcol = wl + (long)(n0 + lmod) * IN_DIM;

  v8f acc = {};
  #pragma unroll
  for (int k0 = 0; k0 < IN_DIM; k0 += 32) {
    // A operand (this code's contiguous k order, same as B): lane-half base k, elems 0-7 = K k..k+7,
    // elems 8-15 = K k+8..k+15
    int ak = k0 + lhi * 16;
    v8bf a0 = *(const v8bf*)(xhrow + ak);
    v8bf a1 = *(const v8bf*)(xhrow + ak + 8);
    v16bf Ah = __builtin_shufflevector(a0, a1, 0,1,2,3,4,5,6,7,8,9,10,11,12,13,14,15);
    v8bf l0 = *(const v8bf*)(xlrow + ak);
    v8bf l1 = *(const v8bf*)(xlrow + ak + 8);
    v16bf Al = __builtin_shufflevector(l0, l1, 0,1,2,3,4,5,6,7,8,9,10,11,12,13,14,15);
    // B operand (16-bit B 32x16): lane=col, 16 consecutive K starting at lhi*16
    int bk = k0 + lhi * 16;
    v16bf Bh = *(const v16bf*)(whcol + bk);
    v16bf Bl = *(const v16bf*)(wlcol + bk);

    acc = __builtin_amdgcn_wmma_f32_16x16x32_bf16(false, Ah, false, Bh, (short)0, acc, false, false);
    acc = __builtin_amdgcn_wmma_f32_16x16x32_bf16(false, Ah, false, Bl, (short)0, acc, false, false);
    acc = __builtin_amdgcn_wmma_f32_16x16x32_bf16(false, Al, false, Bh, (short)0, acc, false, false);
  }
  // C/D layout: VGPR v, lane<16 -> M=v, lane>=16 -> M=v+8; N = lane&15
  #pragma unroll
  for (int v = 0; v < 8; ++v) {
    int row = m0 + v + lhi * 8;
    if (row < Nrows) hout[(long)row * CC + n0 + lmod] = acc[v];
  }
}

// ---------------------------------------------------------------------------
// Node-level attention scalars: s_src/s_dst [N,H]
// ---------------------------------------------------------------------------
__global__ void k_node_scores(const float* __restrict__ h, const float* __restrict__ a_src,
                              const float* __restrict__ a_dst, float* __restrict__ s_src,
                              float* __restrict__ s_dst, int Nn) {
  int i = blockIdx.x * blockDim.x + threadIdx.x;
  if (i >= Nn * HH) return;
  int n = i >> 2, hh = i & 3;
  const float* hp = h + (long)n * CC + hh * DD;
  const float* as = a_src + hh * DD;
  const float* ad = a_dst + hh * DD;
  float ss = 0.f, sd = 0.f;
  for (int d = 0; d < DD; ++d) { float v = hp[d]; ss += v * as[d]; sd += v * ad[d]; }
  s_src[i] = ss; s_dst[i] = sd;
}

__global__ void k_rel_scores(const float* __restrict__ rel_attn, const float* __restrict__ a_rel,
                             float* __restrict__ s_rel, int R) {
  int i = blockIdx.x * blockDim.x + threadIdx.x;
  if (i >= R * HH) return;
  int r = i >> 2, hh = i & 3;
  const float* rp = rel_attn + (long)r * CC + hh * DD;
  const float* ar = a_rel + hh * DD;
  float s = 0.f;
  for (int d = 0; d < DD; ++d) s += rp[d] * ar[d];
  s_rel[i] = s;
}

// ---------------------------------------------------------------------------
// Workspace initialization (harness does not re-poison between replays)
// ---------------------------------------------------------------------------
__global__ void k_init_nh(float* __restrict__ mmax, float* __restrict__ denom, int total) {
  int i = blockIdx.x * blockDim.x + threadIdx.x;
  if (i >= total) return;
  mmax[i] = -INFINITY;
  denom[i] = 0.f;
}

__global__ void k_init_xout(float* __restrict__ xout, const float* __restrict__ bias, long total) {
  long i = (long)blockIdx.x * blockDim.x + threadIdx.x;
  if (i >= total) return;
  xout[i] = bias[i % CC];
}

// ---------------------------------------------------------------------------
// Edge softmax: segment max, then exp + segment sum
// ---------------------------------------------------------------------------
__device__ inline void atomicMaxF(float* addr, float val) {
  unsigned int* ai = (unsigned int*)addr;
  unsigned int old = *ai;
  while (__uint_as_float(old) < val) {
    unsigned int assumed = old;
    old = atomicCAS(ai, assumed, __float_as_uint(val));
    if (old == assumed) break;
  }
}

__device__ inline float edge_logit(const int* ei, const int* et,
                                   const float* s_src, const float* s_dst,
                                   const float* s_rel, int e, int hh, int E, int* dout) {
  int s = ei[e], d = ei[E + e], r = et[e];
  *dout = d;
  float lg = s_src[s * HH + hh] + s_dst[d * HH + hh] + s_rel[r * HH + hh];
  return lg >= 0.f ? lg : NEG_SLOPE * lg;
}

__global__ void k_edge_max(const int* __restrict__ ei, const int* __restrict__ et,
                           const float* __restrict__ s_src, const float* __restrict__ s_dst,
                           const float* __restrict__ s_rel, float* __restrict__ mmax, int E) {
  int i = blockIdx.x * blockDim.x + threadIdx.x;
  if (i >= E * HH) return;
  int e = i >> 2, hh = i & 3, d;
  float lg = edge_logit(ei, et, s_src, s_dst, s_rel, e, hh, E, &d);
  atomicMaxF(&mmax[d * HH + hh], lg);
}

__global__ void k_edge_ex(const int* __restrict__ ei, const int* __restrict__ et,
                          const float* __restrict__ s_src, const float* __restrict__ s_dst,
                          const float* __restrict__ s_rel, const float* __restrict__ mmax,
                          float* __restrict__ exv, float* __restrict__ denom, int E) {
  int i = blockIdx.x * blockDim.x + threadIdx.x;
  if (i >= E * HH) return;
  int e = i >> 2, hh = i & 3, d;
  float lg = edge_logit(ei, et, s_src, s_dst, s_rel, e, hh, E, &d);
  float v = expf(lg - mmax[d * HH + hh]);
  exv[i] = v;
  atomicAdd(&denom[d * HH + hh], v);
}

// ---------------------------------------------------------------------------
// Message scatter: one block per edge, coalesced over the 800-wide feature dim
// ---------------------------------------------------------------------------
__global__ void k_scatter(const int* __restrict__ ei, const float* __restrict__ h,
                          const float* __restrict__ exv, const float* __restrict__ denom,
                          float* __restrict__ xout, int E) {
  int e = blockIdx.x;
  int s = ei[e], d = ei[E + e];
  const float* hs = h + (long)s * CC;
  float* xo = xout + (long)d * CC;
  for (int c = threadIdx.x; c < CC; c += blockDim.x) {
    int hh = c / DD;
    float alpha = exv[(long)e * HH + hh] / (denom[d * HH + hh] + EPSV);
    atomicAdd(&xo[c], hs[c] * alpha);
  }
}

// ---------------------------------------------------------------------------
// DistMult scoring: one block per triple, LDS tree reduction
// ---------------------------------------------------------------------------
__global__ void k_score(const float* __restrict__ x, const float* __restrict__ rel,
                        const int* __restrict__ sidx, const int* __restrict__ ridx,
                        const int* __restrict__ didx, float* __restrict__ out, int B) {
  int b = blockIdx.x;
  const float* xs = x + (long)sidx[b] * CC;
  const float* xd = x + (long)didx[b] * CC;
  const float* rp = rel + (long)ridx[b] * CC;
  __shared__ float red[256];
  float pr = 0.f;
  for (int c = threadIdx.x; c < CC; c += 256) pr += xs[c] * rp[c] * xd[c];
  red[threadIdx.x] = pr;
  __syncthreads();
  for (int s = 128; s > 0; s >>= 1) {
    if (threadIdx.x < s) red[threadIdx.x] += red[threadIdx.x + s];
    __syncthreads();
  }
  if (threadIdx.x == 0) out[b] = red[0];
}

// ---------------------------------------------------------------------------
extern "C" void kernel_launch(void* const* d_in, const int* in_sizes, int n_in,
                              void* d_out, int out_size, void* d_ws, size_t ws_size,
                              hipStream_t stream) {
  const float* node_emb   = (const float*)d_in[0];
  const float* W          = (const float*)d_in[1];
  const float* a_src      = (const float*)d_in[2];
  const float* a_dst      = (const float*)d_in[3];
  const float* a_rel      = (const float*)d_in[4];
  const float* bias       = (const float*)d_in[5];
  const float* rel_attn   = (const float*)d_in[6];
  const float* rel_scorer = (const float*)d_in[7];
  const int*   src_ids    = (const int*)d_in[8];
  const int*   rel_ids    = (const int*)d_in[9];
  const int*   dst_ids    = (const int*)d_in[10];
  const int*   edge_index = (const int*)d_in[11];
  const int*   edge_type  = (const int*)d_in[12];

  const int Nn = in_sizes[0] / IN_DIM;   // 100000
  const int E  = in_sizes[12];           // 400000
  const int B  = in_sizes[8];            // 8192
  const int R  = in_sizes[7] / CC;       // 50

  // Carve workspace (256-byte aligned regions)
  char* p = (char*)d_ws;
  auto alloc = [&](size_t bytes) -> char* {
    char* r = p;
    p += (bytes + 255) & ~(size_t)255;
    return r;
  };
  float*  h     = (float*)alloc((size_t)Nn * CC * 4);
  float*  xout  = (float*)alloc((size_t)Nn * CC * 4);
  __bf16* xh    = (__bf16*)alloc((size_t)Nn * IN_DIM * 2);
  __bf16* xl    = (__bf16*)alloc((size_t)Nn * IN_DIM * 2);
  __bf16* wh    = (__bf16*)alloc((size_t)IN_DIM * CC * 2);
  __bf16* wl    = (__bf16*)alloc((size_t)IN_DIM * CC * 2);
  float*  s_src = (float*)alloc((size_t)Nn * HH * 4);
  float*  s_dst = (float*)alloc((size_t)Nn * HH * 4);
  float*  s_rel = (float*)alloc((size_t)R * HH * 4);
  float*  mmax  = (float*)alloc((size_t)Nn * HH * 4);
  float*  denom = (float*)alloc((size_t)Nn * HH * 4);
  float*  exv   = (float*)alloc((size_t)E * HH * 4);

  long tx = (long)Nn * IN_DIM;
  k_split_x<<<(int)((tx + 255) / 256), 256, 0, stream>>>(node_emb, xh, xl, tx);
  k_split_wT<<<(IN_DIM * CC + 255) / 256, 256, 0, stream>>>(W, wh, wl, IN_DIM, CC);

  int tilesM = (Nn + 15) / 16;
  long waves = (long)tilesM * (CC / 16);
  long thr   = waves * 32;
  k_gemm_wmma<<<(int)((thr + 255) / 256), 256, 0, stream>>>(xh, xl, wh, wl, h, Nn);

  k_node_scores<<<(Nn * HH + 255) / 256, 256, 0, stream>>>(h, a_src, a_dst, s_src, s_dst, Nn);
  k_rel_scores<<<(R * HH + 255) / 256, 256, 0, stream>>>(rel_attn, a_rel, s_rel, R);

  k_init_nh<<<(Nn * HH + 255) / 256, 256, 0, stream>>>(mmax, denom, Nn * HH);
  long txo = (long)Nn * CC;
  k_init_xout<<<(int)((txo + 255) / 256), 256, 0, stream>>>(xout, bias, txo);

  k_edge_max<<<(E * HH + 255) / 256, 256, 0, stream>>>(edge_index, edge_type, s_src, s_dst, s_rel, mmax, E);
  k_edge_ex<<<(E * HH + 255) / 256, 256, 0, stream>>>(edge_index, edge_type, s_src, s_dst, s_rel, mmax, exv, denom, E);

  k_scatter<<<E, 256, 0, stream>>>(edge_index, h, exv, denom, xout, E);

  k_score<<<B, 256, 0, stream>>>(xout, rel_scorer, src_ids, rel_ids, dst_ids, (float*)d_out, B);
}
